// GatedLinearAttention_58660663328939
// MI455X (gfx1250) — matrix operand order repaired and checked by reading, not compiled, not run on hardware
//
#include <hip/hip_runtime.h>
#include <cstdint>

// ---------------------------------------------------------------------------
// Gated Linear Attention forward for MI455X (gfx1250, wave32, WMMA).
// Chunked-parallel GLA; all matmuls via v_wmma_f32_16x16x32_f16; LDS
// fragment loads are 16-byte ds_load_b128; GEMM staging uses the CDNA5
// async global->LDS path (global_load_async_to_lds_b128 + s_wait_asynccnt)
// with double buffering.
// ---------------------------------------------------------------------------

typedef _Float16 h16;
typedef __attribute__((ext_vector_type(16))) _Float16 v16h;
typedef __attribute__((ext_vector_type(8)))  _Float16 v8h;
typedef __attribute__((ext_vector_type(8)))  float    v8f;

#define BATCH 4
#define SEQ   2048
#define DMODEL 1024
#define HEADS 4
#define DKTOT 1024
#define DVTOT 2048
#define DKH   256      // per-head key dim
#define DVH   512      // per-head value dim
#define RLOW  16
#define CHUNK 64       // GLA chunk length
#define NCHUNK (SEQ / CHUNK)
#define DVB   64       // dv slice per workgroup in scan kernel
#define MROWS (BATCH * SEQ)   // 8192

__device__ __forceinline__ v8f wmma_f16(v16h a, v16h b, v8f c) {
  // D = A(16x32 f16) * B(32x16 f16) + C(16x16 f32)
  return __builtin_amdgcn_wmma_f32_16x16x32_f16(false, a, false, b, (short)0, c,
                                                false, false);
}

// Async global->LDS 16B copy (CDNA5 ISA
// op 98). VDST VGPR holds the LDS byte address (generic shared pointer
// truncated to addr[31:0]); VADDR is the 64-bit global address. ASYNCcnt.
__device__ __forceinline__ void async_g2l_b128(uint32_t lds_byte, const void* g) {
  asm volatile("global_load_async_to_lds_b128 %0, %1, off"
               :: "v"(lds_byte), "v"((uint64_t)(uintptr_t)g) : "memory");
}

// --- fragment loaders (CDNA5 WMMA VGPR layouts, CDNA5 ISA) -------
// All loads are 16-byte aligned v8h (-> ds_load_b128).

// A-matrix 16x32 from row-major [M][K]: lane L: row=row0+(L%16),
// K = (L/16)*8 + {0..7, 16..23}
__device__ __forceinline__ v16h load_a_frag(const h16* base, int ld, int row0,
                                            int k0, int lane) {
  const h16* p = base + (size_t)(row0 + (lane & 15)) * ld + k0 + ((lane >> 4) << 3);
  v8h lo = *(const v8h*)p;         // K = kb+0..7
  v8h hi = *(const v8h*)(p + 16);  // K = kb+16..23
  v16h r;
#pragma unroll
  for (int i = 0; i < 8; ++i) { r[i] = lo[i]; r[i + 8] = hi[i]; }
  return r;
}

// B-matrix 32x16 from a K-contiguous source: logical B[kk][n] = src[n][kk]
// lane: N = col0+(L%16); K = (L/16)*8 + {0..7, 16..23}
__device__ __forceinline__ v16h load_bT_frag(const h16* base, int ld, int k0,
                                             int col0, int lane) {
  const h16* p = base + (size_t)(col0 + (lane & 15)) * ld + k0 + ((lane >> 4) << 3);
  v8h lo = *(const v8h*)p;
  v8h hi = *(const v8h*)(p + 16);
  v16h r;
#pragma unroll
  for (int i = 0; i < 8; ++i) { r[i] = lo[i]; r[i + 8] = hi[i]; }
  return r;
}

// --- f32 -> f16 conversion ---------------------------------------------------
__global__ __launch_bounds__(256) void f32_to_f16_k(const float* __restrict__ s,
                                                    h16* __restrict__ d, int n) {
  for (int i = blockIdx.x * 256 + threadIdx.x; i < n; i += gridDim.x * 256)
    d[i] = (h16)s[i];
}

// --- f32 [K][N] -> f16 transposed [N][K] (tiled through LDS) ----------------
__global__ __launch_bounds__(256) void transpose_f16_k(const float* __restrict__ W,
                                                       h16* __restrict__ Wt,
                                                       int K, int Nn) {
  __shared__ float tile[32][33];
  const int k0 = blockIdx.y * 32, n0 = blockIdx.x * 32;
  const int c = threadIdx.x & 31, r = threadIdx.x >> 5;  // 8 rows / pass
#pragma unroll
  for (int rr = 0; rr < 32; rr += 8)
    tile[r + rr][c] = W[(size_t)(k0 + r + rr) * Nn + n0 + c];
  __syncthreads();
#pragma unroll
  for (int rr = 0; rr < 32; rr += 8)
    Wt[(size_t)(n0 + r + rr) * K + k0 + c] = (h16)tile[c][r + rr];
}

// --- generic WMMA GEMM: C[M,N] = alpha * A[M,K] @ Bt[N,K]^T -----------------
// Block 128x128, 256 threads (8 waves, 4x2 wave grid, 32x64 wave tile).
// Double-buffered async global->LDS staging; per K-step each wave does
// 12 ds_load_b128 feeding 8 WMMAs (B-frags shared across both row tiles).
template <bool OUT_HALF>
__global__ __launch_bounds__(256) void gemm_wmma(const h16* __restrict__ A,
                                                 const h16* __restrict__ Bt,
                                                 void* __restrict__ Cout, int M,
                                                 int Nn, int K, float alpha) {
  __shared__ __align__(16) h16 a_s[2][128][32 + 8];
  __shared__ __align__(16) h16 b_s[2][128][32 + 8];
  const int tid = threadIdx.x, lane = tid & 31, w = tid >> 5;
  const int m0 = blockIdx.y * 128, n0 = blockIdx.x * 128;
  const int wr = (w & 3) * 32;                   // two 16-row tiles
  const int wc = (w >> 2) * 64;                  // four 16-col tiles
  const int sr = tid >> 1, sc = (tid & 1) * 16;  // staging coords (16 halves)
  const h16* agp = A  + (size_t)(m0 + sr) * K + sc;
  const h16* bgp = Bt + (size_t)(n0 + sr) * K + sc;
  const int nk = K / 32;
  v8f acc[2][4] = {};

  {  // prologue: async-stage K-tile 0 into buffer 0
    uint32_t al = (uint32_t)(size_t)&a_s[0][sr][sc];
    uint32_t bl = (uint32_t)(size_t)&b_s[0][sr][sc];
    async_g2l_b128(al, agp);      async_g2l_b128(al + 16, agp + 8);
    async_g2l_b128(bl, bgp);      async_g2l_b128(bl + 16, bgp + 8);
  }
  for (int k = 0; k < nk; ++k) {
    if (k + 1 < nk) {  // async-stage next K-tile into the other buffer
      const h16* an = agp + (k + 1) * 32;
      const h16* bn = bgp + (k + 1) * 32;
      uint32_t al = (uint32_t)(size_t)&a_s[(k + 1) & 1][sr][sc];
      uint32_t bl = (uint32_t)(size_t)&b_s[(k + 1) & 1][sr][sc];
      async_g2l_b128(al, an);     async_g2l_b128(al + 16, an + 8);
      async_g2l_b128(bl, bn);     async_g2l_b128(bl + 16, bn + 8);
      asm volatile("s_wait_asynccnt 4" ::: "memory");  // tile k complete
    } else {
      asm volatile("s_wait_asynccnt 0" ::: "memory");
    }
    __syncthreads();
    const h16* ab = &a_s[k & 1][0][0];
    const h16* bb = &b_s[k & 1][0][0];
    v16h af0 = load_a_frag(ab, 40, wr, 0, lane);
    v16h af1 = load_a_frag(ab, 40, wr + 16, 0, lane);
#pragma unroll
    for (int j = 0; j < 4; ++j) {
      v16h bf = load_bT_frag(bb, 40, 0, wc + j * 16, lane);
      acc[0][j] = wmma_f16(af0, bf, acc[0][j]);
      acc[1][j] = wmma_f16(af1, bf, acc[1][j]);
    }
    __syncthreads();
  }
#pragma unroll
  for (int rt = 0; rt < 2; ++rt)
#pragma unroll
    for (int j = 0; j < 4; ++j) {
      int n = n0 + wc + j * 16 + (lane & 15);
#pragma unroll
      for (int v = 0; v < 8; ++v) {
        int m = m0 + wr + rt * 16 + v + ((lane >> 4) << 3);
        float val = acc[rt][j][v] * alpha;
        if (OUT_HALF) ((h16*)Cout)[(size_t)m * Nn + n] = (h16)val;
        else          ((float*)Cout)[(size_t)m * Nn + n] = val;
      }
    }
}

// --- low-rank gate: gk = log_sigmoid((x@Wgk1)@Wgk2 + b) / 16 ----------------
__global__ __launch_bounds__(256) void gate_gk_k(const float* __restrict__ x,
                                                 const float* __restrict__ W1,
                                                 const float* __restrict__ W2,
                                                 const float* __restrict__ bias,
                                                 float* __restrict__ gk) {
  __shared__ float red[256 * RLOW];
  const int row = blockIdx.x, tid = threadIdx.x;
  float part[RLOW];
#pragma unroll
  for (int r = 0; r < RLOW; ++r) part[r] = 0.f;
  for (int d = tid; d < DMODEL; d += 256) {
    float xv = x[(size_t)row * DMODEL + d];
#pragma unroll
    for (int r = 0; r < RLOW; ++r) part[r] += xv * W1[d * RLOW + r];
  }
#pragma unroll
  for (int r = 0; r < RLOW; ++r) red[tid * RLOW + r] = part[r];
  __syncthreads();
  for (int s = 128; s > 0; s >>= 1) {
    if (tid < s)
#pragma unroll
      for (int r = 0; r < RLOW; ++r) red[tid * RLOW + r] += red[(tid + s) * RLOW + r];
    __syncthreads();
  }
  for (int c = tid; c < DKTOT; c += 256) {
    float acc = bias[c];
#pragma unroll
    for (int r = 0; r < RLOW; ++r) acc += red[r] * W2[r * DKTOT + c];
    float ls = fminf(acc, 0.f) - log1pf(__expf(-fabsf(acc)));
    gk[(size_t)row * DKTOT + c] = ls * (1.f / 16.f);
  }
}

// --- chunked GLA scan -------------------------------------------------------
// Grid: B*H*(DVH/DVB)=128 blocks, 256 threads (8 waves). Each block owns a
// 256(dk) x 64(dv) f16 state slice (stored transposed [dv][dk]) in LDS,
// iterates 32 chunks of 64 steps:
//   b_t  = cumsum(gk)            (inclusive, per dk dim)
//   q~_t = q_t * exp(b_t)        k~_t = k_t * exp(-b_t)
//   A    = mask(q~ k~^T)         o = A v + q~ S_prev
//   S    = exp(b_C) ⊙ (S + k~^T v)
__global__ __launch_bounds__(256) void gla_scan_k(const h16* __restrict__ qh,
                                                  const h16* __restrict__ kh,
                                                  const h16* __restrict__ vh,
                                                  const float* __restrict__ gk,
                                                  float* __restrict__ of) {
  extern __shared__ __align__(16) char smem[];
  h16*   ShT = (h16*)smem;                  // [64][264]  state^T (dv x dk), f16
  h16*   qt  = (h16*)(smem + 33792);        // [64][264]  q~ chunk (t x dk)
  h16*   kt  = (h16*)(smem + 67584);        // [64][264]  k~ chunk (t x dk)
  h16*   ktT = (h16*)(smem + 101376);       // [256][72]  k~^T (dk x t)
  h16*   vtT = (h16*)(smem + 138240);       // [64][72]   v^T (dv x t)
  h16*   Ah  = (h16*)(smem + 147456);       // [64][72]   masked attention, f16
  float* eB  = (float*)(smem + 156672);     // [256]      exp(chunk-total decay)

  const int tid = threadIdx.x, lane = tid & 31, w = tid >> 5;
  const int bh = blockIdx.x >> 3, slice = blockIdx.x & 7;
  const int b = bh >> 2, hh = bh & 3;
  const int dv0 = slice * DVB;

  for (int i = tid; i < 64 * 264; i += 256) ShT[i] = (h16)0.f;
  __syncthreads();

  const int tr = (w >> 1) * 16;   // query tile row for phases 2/3
  const int tc = (w & 1) * 32;    // two 16-col tiles

  for (int nc = 0; nc < NCHUNK; ++nc) {
    const int t0 = nc * CHUNK;
    const int row_base = b * SEQ + t0;

    // ---- phase 1: decay-weighted q/k staging (thread = dk column) ----------
    {
      const int d = tid;
      const float* gp = gk + (size_t)row_base * DKTOT + hh * DKH + d;
      const h16*   qp = qh + (size_t)row_base * DKTOT + hh * DKH + d;
      const h16*   kp = kh + (size_t)row_base * DKTOT + hh * DKH + d;
      float cum = 0.f;
      for (int tb = 0; tb < CHUNK; tb += 8) {
        v8h kbuf;
#pragma unroll
        for (int u = 0; u < 8; ++u) {
          int t = tb + u;
          cum += gp[(size_t)t * DKTOT];
          h16 qv = (h16)((float)qp[(size_t)t * DKTOT] * __expf(cum));
          h16 kv = (h16)((float)kp[(size_t)t * DKTOT] * __expf(-cum));
          qt[t * 264 + d] = qv;
          kt[t * 264 + d] = kv;
          kbuf[u] = kv;
        }
        *(v8h*)&ktT[d * 72 + tb] = kbuf;   // transposed copy, b128 store
      }
      eB[d] = __expf(cum);
      for (int i = tid; i < CHUNK * DVB; i += 256) {  // v^T staging
        int t = i >> 6, j = i & 63;
        vtT[j * 72 + t] = vh[(size_t)(row_base + t) * DVTOT + hh * DVH + dv0 + j];
      }
    }
    __syncthreads();

    // ---- phase 2: o_inter = q~ @ S_prev ; A = q~ @ k~^T (shared A-frags) ---
    v8f oacc[2] = {};
    v8f aacc[2] = {};
    for (int k0 = 0; k0 < DKH; k0 += 32) {
      v16h aq = load_a_frag(qt, 264, tr, k0, lane);
#pragma unroll
      for (int j = 0; j < 2; ++j) {
        oacc[j] = wmma_f16(aq, load_bT_frag(ShT, 264, k0, tc + j * 16, lane), oacc[j]);
        aacc[j] = wmma_f16(aq, load_bT_frag(kt,  264, k0, tc + j * 16, lane), aacc[j]);
      }
    }
#pragma unroll
    for (int j = 0; j < 2; ++j) {  // causal mask, convert to f16
      int s = tc + j * 16 + (lane & 15);
#pragma unroll
      for (int v = 0; v < 8; ++v) {
        int t = tr + v + ((lane >> 4) << 3);
        Ah[t * 72 + s] = (h16)((s <= t) ? aacc[j][v] : 0.f);
      }
    }
    __syncthreads();

    // ---- phase 3: o += A @ v; write o --------------------------------------
    for (int k0 = 0; k0 < CHUNK; k0 += 32) {
      v16h aa = load_a_frag(Ah, 72, tr, k0, lane);
#pragma unroll
      for (int j = 0; j < 2; ++j)
        oacc[j] = wmma_f16(aa, load_bT_frag(vtT, 72, k0, tc + j * 16, lane), oacc[j]);
    }
#pragma unroll
    for (int j = 0; j < 2; ++j) {
      int n = tc + j * 16 + (lane & 15);
#pragma unroll
      for (int v = 0; v < 8; ++v) {
        int t = tr + v + ((lane >> 4) << 3);
        of[(size_t)(row_base + t) * DVTOT + hh * DVH + dv0 + n] = oacc[j][v];
      }
    }
    __syncthreads();

    // ---- phase 4: S = exp(B) ⊙ (S + k~^T @ v), vectorized RMW on ShT -------
#pragma unroll
    for (int rr = 0; rr < 2; ++rr) {
      int dr = (w * 2 + rr) * 16;  // dk tile base, 8 waves cover 256
      v8f uacc[4] = {};
      for (int k0 = 0; k0 < CHUNK; k0 += 32) {
        v16h ak = load_a_frag(ktT, 72, dr, k0, lane);
#pragma unroll
        for (int j = 0; j < 4; ++j)
          uacc[j] = wmma_f16(ak, load_bT_frag(vtT, 72, k0, j * 16, lane), uacc[j]);
      }
#pragma unroll
      for (int j = 0; j < 4; ++j) {
        int n  = j * 16 + (lane & 15);
        int m0 = dr + ((lane >> 4) << 3);     // 8 contiguous dk per lane
        v8h sold = *(v8h*)&ShT[n * 264 + m0];
        v8h snew;
#pragma unroll
        for (int v = 0; v < 8; ++v)
          snew[v] = (h16)(((float)sold[v] + uacc[j][v]) * eB[m0 + v]);
        *(v8h*)&ShT[n * 264 + m0] = snew;
      }
    }
    __syncthreads();
  }
}

// --- fused RMSNorm + swish gate --------------------------------------------
__global__ __launch_bounds__(256) void norm_gate_k(const float* __restrict__ of,
                                                   const h16* __restrict__ gf,
                                                   const float* __restrict__ wn,
                                                   h16* __restrict__ onh) {
  __shared__ float red[256];
  const int tid = threadIdx.x;
  const int row = blockIdx.x / HEADS, h = blockIdx.x % HEADS;
  const size_t base = (size_t)row * DVTOT + h * DVH;
  float v0 = of[base + tid], v1 = of[base + tid + 256];
  red[tid] = v0 * v0 + v1 * v1;
  __syncthreads();
  for (int s = 128; s > 0; s >>= 1) {
    if (tid < s) red[tid] += red[tid + s];
    __syncthreads();
  }
  float r = rsqrtf(red[0] * (1.f / DVH) + 1e-5f);
  float g0 = (float)gf[base + tid], g1 = (float)gf[base + tid + 256];
  float s0 = g0 / (1.f + __expf(-g0)), s1 = g1 / (1.f + __expf(-g1));
  onh[base + tid]       = (h16)(v0 * r * wn[tid] * s0);
  onh[base + tid + 256] = (h16)(v1 * r * wn[tid + 256] * s1);
}

// ---------------------------------------------------------------------------
extern "C" void kernel_launch(void* const* d_in, const int* in_sizes, int n_in,
                              void* d_out, int out_size, void* d_ws, size_t ws_size,
                              hipStream_t stream) {
  const float* x    = (const float*)d_in[0];
  const float* Wq   = (const float*)d_in[1];
  const float* Wk   = (const float*)d_in[2];
  const float* Wv   = (const float*)d_in[3];
  const float* Wg   = (const float*)d_in[4];
  const float* Wgk1 = (const float*)d_in[5];
  const float* Wgk2 = (const float*)d_in[6];
  const float* bgk2 = (const float*)d_in[7];
  const float* Wo   = (const float*)d_in[8];
  const float* gnw  = (const float*)d_in[9];
  float* out = (float*)d_out;

  char* ws = (char*)d_ws;
  size_t off = 0;
  auto alloc = [&](size_t bytes) { char* p = ws + off; off = (off + bytes + 255) & ~size_t(255); return p; };
  h16*   xh   = (h16*)alloc((size_t)MROWS * DMODEL * 2);
  h16*   WqT  = (h16*)alloc((size_t)DMODEL * DKTOT * 2);   // [N][K]
  h16*   WkT  = (h16*)alloc((size_t)DMODEL * DKTOT * 2);
  h16*   WvT  = (h16*)alloc((size_t)DMODEL * DVTOT * 2);
  h16*   WgT  = (h16*)alloc((size_t)DMODEL * DVTOT * 2);
  h16*   WoT  = (h16*)alloc((size_t)DVTOT * DMODEL * 2);
  h16*   qh   = (h16*)alloc((size_t)MROWS * DKTOT * 2);
  h16*   kh   = (h16*)alloc((size_t)MROWS * DKTOT * 2);
  h16*   vh   = (h16*)alloc((size_t)MROWS * DVTOT * 2);
  h16*   gf   = (h16*)alloc((size_t)MROWS * DVTOT * 2);
  float* gkb  = (float*)alloc((size_t)MROWS * DKTOT * 4);
  float* ofb  = (float*)alloc((size_t)MROWS * DVTOT * 4);
  h16*   onh  = (h16*)alloc((size_t)MROWS * DVTOT * 2);

  // 1) convert x; transpose+convert weights to [N][K] f16
  f32_to_f16_k<<<2048, 256, 0, stream>>>(x, xh, MROWS * DMODEL);
  transpose_f16_k<<<dim3(DKTOT / 32, DMODEL / 32), 256, 0, stream>>>(Wq, WqT, DMODEL, DKTOT);
  transpose_f16_k<<<dim3(DKTOT / 32, DMODEL / 32), 256, 0, stream>>>(Wk, WkT, DMODEL, DKTOT);
  transpose_f16_k<<<dim3(DVTOT / 32, DMODEL / 32), 256, 0, stream>>>(Wv, WvT, DMODEL, DVTOT);
  transpose_f16_k<<<dim3(DVTOT / 32, DMODEL / 32), 256, 0, stream>>>(Wg, WgT, DMODEL, DVTOT);
  transpose_f16_k<<<dim3(DMODEL / 32, DVTOT / 32), 256, 0, stream>>>(Wo, WoT, DVTOT, DMODEL);

  // 2) projections (q scaled by dk^-0.5 = 1/16)
  gemm_wmma<true><<<dim3(DKTOT / 128, MROWS / 128), 256, 0, stream>>>(xh, WqT, qh, MROWS, DKTOT, DMODEL, 0.0625f);
  gemm_wmma<true><<<dim3(DKTOT / 128, MROWS / 128), 256, 0, stream>>>(xh, WkT, kh, MROWS, DKTOT, DMODEL, 1.f);
  gemm_wmma<true><<<dim3(DVTOT / 128, MROWS / 128), 256, 0, stream>>>(xh, WvT, vh, MROWS, DVTOT, DMODEL, 1.f);
  gemm_wmma<true><<<dim3(DVTOT / 128, MROWS / 128), 256, 0, stream>>>(xh, WgT, gf, MROWS, DVTOT, DMODEL, 1.f);

  // 3) low-rank gate logits -> log-decay
  gate_gk_k<<<MROWS, 256, 0, stream>>>(x, Wgk1, Wgk2, bgk2, gkb);

  // 4) chunked GLA scan (dynamic LDS ≈ 154 KB / workgroup)
  gla_scan_k<<<BATCH * HEADS * (DVH / DVB), 256, 157696, stream>>>(qh, kh, vh, gkb, ofb);

  // 5) RMSNorm + swish gate
  norm_gate_k<<<MROWS * HEADS, 256, 0, stream>>>(ofb, gf, gnw, onh);

  // 6) output projection -> d_out (f32)
  gemm_wmma<false><<<dim3(DMODEL / 128, MROWS / 128), 256, 0, stream>>>(onh, WoT, out, MROWS, DMODEL, DVTOT, 1.f);

  (void)in_sizes; (void)n_in; (void)out_size; (void)ws_size;
}
